// Net_73718818668739
// MI455X (gfx1250) — matrix operand order repaired and checked by reading, not compiled, not run on hardware
//
#include <hip/hip_runtime.h>
#include <hip/hip_bf16.h>
#include <math.h>

typedef __attribute__((ext_vector_type(16))) _Float16 v16h;
typedef __attribute__((ext_vector_type(8)))  float    v8f;

// ---------------------------------------------------------------------------
// Degree / normalization
// ---------------------------------------------------------------------------
__global__ __launch_bounds__(256) void deg_init(float* __restrict__ deg, int n) {
    int i = blockIdx.x * 256 + threadIdx.x;
    if (i < n) deg[i] = 1.0f;                      // self-loop contributes 1
}

__global__ __launch_bounds__(256) void deg_edges(const int* __restrict__ dst,
                                                 float* __restrict__ deg, int E) {
    int e = blockIdx.x * 256 + threadIdx.x;
    if (e < E) atomicAdd(&deg[dst[e]], 1.0f);
}

__global__ __launch_bounds__(256) void deg_rsqrt(float* __restrict__ deg, int n) {
    int i = blockIdx.x * 256 + threadIdx.x;
    if (i < n) deg[i] = rsqrtf(deg[i]);            // deg >= 1 always (self-loop)
}

// ---------------------------------------------------------------------------
// WMMA index helpers (CDNA5 ISA §7.12.2, wave32)
// A 16x32 f16: lane&15 = row M; half j -> K:
//   lane<16 : j<8 -> j      ; j>=8 -> j+8
//   lane>=16: j<8 -> j+8    ; j>=8 -> j+16
// B 32x16 f16: lane&15 = col N; half j -> K same as A (j + (j>=8 ? 8 : 0) + (lane>=16 ? 8 : 0))
// C/D 16x16 f32: VGPR v -> M = v + (lane>=16 ? 8 : 0); N = lane&15
// ---------------------------------------------------------------------------
__device__ __forceinline__ int a16_k(int lane, int j) {
    return j + ((j >= 8) ? 8 : 0) + ((lane >= 16) ? 8 : 0);
}
__device__ __forceinline__ int b16_k(int lane, int j) {
    return j + ((j >= 8) ? 8 : 0) + ((lane >= 16) ? 8 : 0);
}

// ---------------------------------------------------------------------------
// Layer 1 transform: xw1[.][32] = x[.][18] @ W1[18][32]   (K padded 18 -> 32)
// FULL tiles only: no bounds checks anywhere -> unconditional vmem, full EXEC.
// K-padding via clamped load + select (v_cndmask), not branches.
// ---------------------------------------------------------------------------
__global__ __launch_bounds__(256) void xw1_wmma(const float* __restrict__ x,
                                                const float* __restrict__ W1,
                                                float* __restrict__ xw1,
                                                int nTiles) {
    const int lane = threadIdx.x & 31;
    const int tile = blockIdx.x * 8 + (threadIdx.x >> 5);
    if (tile >= nTiles) return;                    // wave-uniform: EXEC stays all-1
    const int row0 = tile * 16;
    const int m    = lane & 15;
    const int arow = row0 + m;                     // always in range (full tile)

    v16h a;
#pragma unroll
    for (int j = 0; j < 16; ++j) {
        int k  = a16_k(lane, j);
        int kc = (k < 18) ? k : 17;                // clamp: load always executes
        float v = x[arow * 18 + kc];
        a[j] = (_Float16)((k < 18) ? v : 0.0f);    // select, no branch
    }

#pragma unroll
    for (int t = 0; t < 2; ++t) {                  // N-tiles: cols [0,16), [16,32)
        v16h b;
#pragma unroll
        for (int j = 0; j < 16; ++j) {
            int k  = b16_k(lane, j);
            int kc = (k < 18) ? k : 17;
            float v = W1[kc * 32 + (t * 16 + m)];
            b[j] = (_Float16)((k < 18) ? v : 0.0f);
        }
        v8f c = {};
        c = __builtin_amdgcn_wmma_f32_16x16x32_f16(false, a, false, b,
                                                   (short)0, c, false, false);
#pragma unroll
        for (int v = 0; v < 8; ++v) {
            int M = v + ((lane >= 16) ? 8 : 0);
            xw1[(row0 + M) * 32 + t * 16 + m] = c[v];   // unconditional store
        }
    }
}

// ---------------------------------------------------------------------------
// Layer 2 transform: xw2[.][64] = relu(agg1[.][32] + b1) @ W2[32][64]
// Bias + ReLU of layer 1 fused into the A-operand load. K = 32 exactly.
// ---------------------------------------------------------------------------
__global__ __launch_bounds__(256) void xw2_wmma(const float* __restrict__ agg1,
                                                const float* __restrict__ b1,
                                                const float* __restrict__ W2,
                                                float* __restrict__ xw2,
                                                int nTiles) {
    const int lane = threadIdx.x & 31;
    const int tile = blockIdx.x * 8 + (threadIdx.x >> 5);
    if (tile >= nTiles) return;
    const int row0 = tile * 16;
    const int m    = lane & 15;
    const int arow = row0 + m;

    v16h a;
#pragma unroll
    for (int j = 0; j < 16; ++j) {
        int k = a16_k(lane, j);
        float h = agg1[arow * 32 + k] + b1[k];
        a[j] = (_Float16)fmaxf(h, 0.0f);           // fused bias + ReLU
    }

#pragma unroll
    for (int t = 0; t < 4; ++t) {                  // N-tiles: 4 x 16 = 64 cols
        v16h b;
#pragma unroll
        for (int j = 0; j < 16; ++j) {
            int k = b16_k(lane, j);
            b[j] = (_Float16)W2[k * 64 + (t * 16 + m)];
        }
        v8f c = {};
        c = __builtin_amdgcn_wmma_f32_16x16x32_f16(false, a, false, b,
                                                   (short)0, c, false, false);
#pragma unroll
        for (int v = 0; v < 8; ++v) {
            int M = v + ((lane >= 16) ? 8 : 0);
            xw2[(row0 + M) * 64 + t * 16 + m] = c[v];
        }
    }
}

// ---------------------------------------------------------------------------
// Scalar tail for rows not covered by full 16-row tiles (launched only when
// N % 16 != 0; for N=100000 it never runs). KIN doubles as the input stride.
// ---------------------------------------------------------------------------
template <int KIN, int NOUT, bool RELU>
__global__ __launch_bounds__(256) void gemm_tail(const float* __restrict__ in,
                                                 const float* __restrict__ bias,
                                                 const float* __restrict__ W,
                                                 float* __restrict__ out,
                                                 int rowStart, int n) {
    int idx = blockIdx.x * 256 + threadIdx.x;
    int total = (n - rowStart) * NOUT;
    if (idx >= total) return;
    int r = rowStart + idx / NOUT;
    int c = idx % NOUT;
    float acc = 0.0f;
#pragma unroll
    for (int k = 0; k < KIN; ++k) {
        float v = in[r * KIN + k];
        if (RELU) v = fmaxf(v + bias[k], 0.0f);
        acc += v * W[k * NOUT + c];
    }
    out[r * NOUT + c] = acc;
}

// ---------------------------------------------------------------------------
// Aggregation. Stream-serialized fusion: the init kernel writes the self-loop
// term (xw[i]*dinv[i]^2) directly -- doubles as the zero-init -- then the edge
// kernel atomically accumulates the 1.28M edge messages (hits 192MB L2).
// ---------------------------------------------------------------------------
template <int F>
__global__ __launch_bounds__(256) void agg_selfloop_init(const float* __restrict__ xw,
                                                         const float* __restrict__ dinv,
                                                         float* __restrict__ out,
                                                         int n) {
    unsigned idx = blockIdx.x * 256u + threadIdx.x;
    unsigned total = (unsigned)n * F;
    if (idx >= total) return;
    int i = idx / F;
    float di = dinv[i];
    out[idx] = xw[idx] * di * di;
}

template <int F, int SHIFT>
__global__ __launch_bounds__(256) void agg_edges(const float* __restrict__ xw,
                                                 const int* __restrict__ src,
                                                 const int* __restrict__ dst,
                                                 const float* __restrict__ dinv,
                                                 float* __restrict__ out,
                                                 int E) {
    unsigned idx = blockIdx.x * 256u + threadIdx.x;
    if (idx >= ((unsigned)E << SHIFT)) return;
    int e = idx >> SHIFT;
    int f = idx & (F - 1);
    int s = src[e];
    int d = dst[e];
    float nrm = dinv[s] * dinv[d];
    atomicAdd(&out[d * F + f], xw[s * F + f] * nrm);   // global_atomic_add_f32
}

// ---------------------------------------------------------------------------
// Head: h2 = relu(agg2 + b2); logits = h2 @ Wfc + bfc; log_softmax(2).
// One wave per node, wave32 shuffle reduction.
// ---------------------------------------------------------------------------
__global__ __launch_bounds__(256) void head_kernel(const float* __restrict__ agg2,
                                                   const float* __restrict__ b2,
                                                   const float* __restrict__ Wfc,
                                                   const float* __restrict__ bfc,
                                                   float* __restrict__ out, int n) {
    const int lane = threadIdx.x & 31;
    const int node = blockIdx.x * 8 + (threadIdx.x >> 5);
    if (node >= n) return;
    float ha = fmaxf(agg2[node * 64 + lane]      + b2[lane],      0.0f);
    float hb = fmaxf(agg2[node * 64 + 32 + lane] + b2[32 + lane], 0.0f);
    float p0 = ha * Wfc[lane * 2 + 0] + hb * Wfc[(32 + lane) * 2 + 0];
    float p1 = ha * Wfc[lane * 2 + 1] + hb * Wfc[(32 + lane) * 2 + 1];
#pragma unroll
    for (int off = 16; off > 0; off >>= 1) {
        p0 += __shfl_xor(p0, off, 32);
        p1 += __shfl_xor(p1, off, 32);
    }
    if (lane == 0) {
        float l0 = p0 + bfc[0];
        float l1 = p1 + bfc[1];
        float mx  = fmaxf(l0, l1);
        float lse = mx + logf(expf(l0 - mx) + expf(l1 - mx));
        out[node * 2 + 0] = l0 - lse;
        out[node * 2 + 1] = l1 - lse;
    }
}

// ---------------------------------------------------------------------------
// Launch
// ---------------------------------------------------------------------------
extern "C" void kernel_launch(void* const* d_in, const int* in_sizes, int n_in,
                              void* d_out, int out_size, void* d_ws, size_t ws_size,
                              hipStream_t stream) {
    const float* x   = (const float*)d_in[0];
    const int*   ei  = (const int*)  d_in[1];
    const float* W1  = (const float*)d_in[2];
    const float* b1  = (const float*)d_in[3];
    const float* W2  = (const float*)d_in[4];
    const float* b2  = (const float*)d_in[5];
    const float* Wfc = (const float*)d_in[6];
    const float* bfc = (const float*)d_in[7];
    float* out = (float*)d_out;

    const int N = in_sizes[0] / 18;
    const int E = in_sizes[1] / 2;
    const int* src = ei;
    const int* dst = ei + E;

    // Workspace layout (floats): dinv | xw1 | agg1 | xw2 | agg2  (~77 MB)
    float* ws   = (float*)d_ws;
    float* dinv = ws;
    float* xw1  = ws   + (((size_t)N + 255) & ~(size_t)255);
    float* agg1 = xw1  + (size_t)N * 32;
    float* xw2  = agg1 + (size_t)N * 32;
    float* agg2 = xw2  + (size_t)N * 64;

    const int nFull  = N / 16;                     // full WMMA tiles
    const int tail0  = nFull * 16;                 // first tail row
    const int nTail  = N - tail0;                  // 0 for N=100000

    const unsigned bN   = (N + 255) / 256;
    const unsigned bE   = (E + 255) / 256;
    const unsigned bT   = (nFull + 7) / 8;
    const unsigned bA1i = ((unsigned)N * 32u + 255u) / 256u;
    const unsigned bA1e = ((unsigned)E * 32u + 255u) / 256u;
    const unsigned bA2i = ((unsigned)N * 64u + 255u) / 256u;
    const unsigned bA2e = ((unsigned)E * 64u + 255u) / 256u;
    const unsigned bH   = (N + 7) / 8;

    // deg (with self-loops) -> dinv
    deg_init <<<bN, 256, 0, stream>>>(dinv, N);
    deg_edges<<<bE, 256, 0, stream>>>(dst, dinv, E);
    deg_rsqrt<<<bN, 256, 0, stream>>>(dinv, N);

    // layer 1: transform (WMMA, full tiles) + scalar tail -> self-loop -> edges
    if (nFull > 0)
        xw1_wmma<<<bT, 256, 0, stream>>>(x, W1, xw1, nFull);
    if (nTail > 0)
        gemm_tail<18, 32, false><<<(nTail * 32 + 255) / 256, 256, 0, stream>>>(
            x, nullptr, W1, xw1, tail0, N);
    agg_selfloop_init<32><<<bA1i, 256, 0, stream>>>(xw1, dinv, agg1, N);
    agg_edges<32, 5>    <<<bA1e, 256, 0, stream>>>(xw1, src, dst, dinv, agg1, E);

    // layer 2: (bias+ReLU fused) transform (WMMA) + tail -> self-loop -> edges
    if (nFull > 0)
        xw2_wmma<<<bT, 256, 0, stream>>>(agg1, b1, W2, xw2, nFull);
    if (nTail > 0)
        gemm_tail<32, 64, true><<<(nTail * 64 + 255) / 256, 256, 0, stream>>>(
            agg1, b1, W2, xw2, tail0, N);
    agg_selfloop_init<64><<<bA2i, 256, 0, stream>>>(xw2, dinv, agg2, N);
    agg_edges<64, 6>    <<<bA2e, 256, 0, stream>>>(xw2, src, dst, dinv, agg2, E);

    // head: bias+ReLU + FC + log_softmax
    head_kernel<<<bH, 256, 0, stream>>>(agg2, b2, Wfc, bfc, out, N);
}
